// MultiHeadCrossAttention_86930138071604
// MI455X (gfx1250) — compile-verified, matrix operand order checked by reading, not run on hardware
//
#include <hip/hip_runtime.h>

// ---------------------------------------------------------------------------
// Multi-head cross attention on MI455X (gfx1250, wave32, WMMA, TDM).
// cast f32->bf16 -> GEMM Q,K,V -> flash attention (online softmax) -> GEMM+bias
// All LDS double-buffered; attention K chunks DMA'd by the Tensor Data Mover
// overlapped with compute (tensor_load_to_lds + s_wait_tensorcnt).
// ---------------------------------------------------------------------------

typedef __attribute__((ext_vector_type(16))) __bf16 v16bf;
typedef __attribute__((ext_vector_type(8)))  float  v8f;
typedef __attribute__((ext_vector_type(4)))  unsigned int u32x4_t;
typedef __attribute__((ext_vector_type(8)))  int          i32x8_t;
typedef __attribute__((ext_vector_type(4)))  int          i32x4_t;

union FragBF {
    uint4  u[2];
    v16bf  v;
};
union Pack8 {
    uint4  u;
    __bf16 e[8];
};

#define BATCH      4
#define NQ         4096
#define MK         1024
#define EMB        1024
#define CTX        768
#define NHEAD      16
#define HDIM       64
#define ATT_SCALE  0.125f   // 64^-0.5

#if __has_builtin(__builtin_amdgcn_tensor_load_to_lds)
#define HAVE_TDM 1
#else
#define HAVE_TDM 0
#endif

static __device__ __forceinline__ v8f wmma_bf16(v16bf a, v16bf b, v8f c) {
    return __builtin_amdgcn_wmma_f32_16x16x32_bf16(false, a, false, b, (short)0, c,
                                                   false, false);
}

#if HAVE_TDM
// TDM 2D tile load: rows x width bf16, global row stride row_stride (elements),
// packed row-major into LDS at lds_off. Completion via s_wait_tensorcnt.
static __device__ __forceinline__ void tdm_load_2d_bf16(const __bf16* gbase,
                                                        unsigned lds_off, int rows,
                                                        int width, int row_stride) {
    unsigned long long ga = (unsigned long long)(uintptr_t)gbase;
    u32x4_t g0;
    g0[0] = 1u;                                                 // count=1
    g0[1] = lds_off;                                            // LDS byte address
    g0[2] = (unsigned)(ga & 0xFFFFFFFFu);                       // global_addr lo
    g0[3] = (unsigned)((ga >> 32) & 0x01FFFFFFu) | (2u << 30);  // hi | type=2
    i32x8_t g1;
    g1[0] = (int)(1u << 16);               // mask=0, data_size=1 (2B)
    g1[1] = (int)((unsigned)width << 16);  // atomic_addr=0 | tensor_dim0 lo16
    g1[2] = (int)((unsigned)rows << 16);   // tensor_dim0 hi=0 | tensor_dim1 lo16
    g1[3] = (int)((unsigned)width << 16);  // tensor_dim1 hi=0 | tile_dim0
    g1[4] = (int)(unsigned)rows;           // tile_dim1 | tile_dim2=0
    g1[5] = row_stride;                    // tensor_dim0_stride[31:0]
    g1[6] = 0;
    g1[7] = 0;
    i32x4_t z4 = {0, 0, 0, 0};
#if defined(__clang_major__) && __clang_major__ >= 23
    i32x8_t z8 = {0, 0, 0, 0, 0, 0, 0, 0};
    __builtin_amdgcn_tensor_load_to_lds(g0, g1, z4, z4, z8, 0);
#else
    __builtin_amdgcn_tensor_load_to_lds(g0, g1, z4, z4, 0);
#endif
}
#endif

// ---------------------------------------------------------------------------
__global__ __launch_bounds__(256) void cast_f32_bf16(const float* __restrict__ src,
                                                     __bf16* __restrict__ dst, int n) {
    int i      = blockIdx.x * blockDim.x + threadIdx.x;
    int stride = gridDim.x * blockDim.x;
    int n4     = n >> 2;
    for (; i < n4; i += stride) {
        float4 v = ((const float4*)src)[i];
        dst[4 * i + 0] = (__bf16)v.x;
        dst[4 * i + 1] = (__bf16)v.y;
        dst[4 * i + 2] = (__bf16)v.z;
        dst[4 * i + 3] = (__bf16)v.w;
    }
}

// ---------------------------------------------------------------------------
// GEMM: C[M,N] = A[M,K] @ B[K,N] (+bias). Block 256 thr = 8 waves (2x4),
// block tile 128x128, wave tile 64x32 (4x2 accumulators). Double-buffered LDS,
// one barrier per k-step; epilogue fully resolved at compile time.
// Requires M%128==0, N%128==0, K%32==0.
// ---------------------------------------------------------------------------
template <bool WF32, bool WBF16, bool BIAS>
__global__ __launch_bounds__(256) void gemm_bf16_wmma(
    const __bf16* __restrict__ A, const __bf16* __restrict__ Bm,
    const float* __restrict__ bias, float* __restrict__ Cf,
    __bf16* __restrict__ Cbf, int Mdim, int Ndim, int Kdim) {
    __shared__ __bf16 sA[2][128][32];   // [buf][row][k]
    __shared__ __bf16 sBt[2][128][32];  // [buf][col][k]

    const int tid   = threadIdx.x;
    const int wave  = tid >> 5;
    const int lane  = tid & 31;
    const int lrow  = lane & 15;
    const int koff  = (lane >> 4) << 3;  // 0 or 8
    const int wr    = wave >> 2;         // M offset 64*wr
    const int wc    = wave & 3;          // N offset 32*wc
    const int rbase = blockIdx.y * 128;
    const int cbase = blockIdx.x * 128;

    const int arow = tid >> 2;        // 0..63 (and +64)
    const int ak   = (tid & 3) * 8;   // 0,8,16,24
    const int bk   = tid >> 4;        // 0..15 (and +16)
    const int bc   = (tid & 15) * 8;  // 0..120

    const __bf16* Ap0 = A + (size_t)(rbase + arow) * Kdim + ak;
    const __bf16* Ap1 = A + (size_t)(rbase + arow + 64) * Kdim + ak;
    const __bf16* Bp0 = Bm + (size_t)bk * Ndim + cbase + bc;
    const __bf16* Bp1 = Bm + (size_t)(bk + 16) * Ndim + cbase + bc;

    v8f acc[4][2];
#pragma unroll
    for (int i = 0; i < 4; ++i) {
        acc[i][0] = (v8f){};
        acc[i][1] = (v8f){};
    }

    // ---- prologue: stage k-step 0 into buffer 0 ----
    {
        uint4 a0 = *(const uint4*)(Ap0);
        uint4 a1 = *(const uint4*)(Ap1);
        Pack8 b0, b1;
        b0.u = *(const uint4*)(Bp0);
        b1.u = *(const uint4*)(Bp1);
        *(uint4*)(&sA[0][arow][ak])      = a0;
        *(uint4*)(&sA[0][arow + 64][ak]) = a1;
#pragma unroll
        for (int j = 0; j < 8; ++j) {
            sBt[0][bc + j][bk]      = b0.e[j];
            sBt[0][bc + j][bk + 16] = b1.e[j];
        }
    }
    __syncthreads();

    const int nsteps = Kdim >> 5;
    for (int s = 0; s < nsteps; ++s) {
        const int  buf      = s & 1;
        const bool has_next = (s + 1) < nsteps;
        uint4 a0 = {}, a1 = {};
        Pack8 b0 = {}, b1 = {};
        if (has_next) {  // issue next tile's global loads early
            const int kk = (s + 1) * 32;
            a0   = *(const uint4*)(Ap0 + kk);
            a1   = *(const uint4*)(Ap1 + kk);
            b0.u = *(const uint4*)(Bp0 + (size_t)kk * Ndim);
            b1.u = *(const uint4*)(Bp1 + (size_t)kk * Ndim);
        }
        // ---- compute on current buffer ----
        FragBF bf[2];
#pragma unroll
        for (int t = 0; t < 2; ++t) {
            bf[t].u[0] = *(const uint4*)(&sBt[buf][wc * 32 + 16 * t + lrow][koff]);
            bf[t].u[1] = *(const uint4*)(&sBt[buf][wc * 32 + 16 * t + lrow][16 + koff]);
        }
#pragma unroll
        for (int i = 0; i < 4; ++i) {
            FragBF af;
            af.u[0] = *(const uint4*)(&sA[buf][wr * 64 + 16 * i + lrow][koff]);
            af.u[1] = *(const uint4*)(&sA[buf][wr * 64 + 16 * i + lrow][16 + koff]);
            acc[i][0] = wmma_bf16(af.v, bf[0].v, acc[i][0]);
            acc[i][1] = wmma_bf16(af.v, bf[1].v, acc[i][1]);
        }
        // ---- fill the other buffer ----
        if (has_next) {
            *(uint4*)(&sA[buf ^ 1][arow][ak])      = a0;
            *(uint4*)(&sA[buf ^ 1][arow + 64][ak]) = a1;
#pragma unroll
            for (int j = 0; j < 8; ++j) {
                sBt[buf ^ 1][bc + j][bk]      = b0.e[j];
                sBt[buf ^ 1][bc + j][bk + 16] = b1.e[j];
            }
        }
        __syncthreads();
    }

    // ---- epilogue (compile-time specialization, branch-free) ----
#pragma unroll
    for (int t = 0; t < 2; ++t) {
        const int col = cbase + wc * 32 + 16 * t + lrow;
        float bv = 0.0f;
        if constexpr (BIAS) bv = bias[col];
#pragma unroll
        for (int i = 0; i < 4; ++i) {
#pragma unroll
            for (int r = 0; r < 8; ++r) {
                const size_t row = (size_t)(rbase + wr * 64 + 16 * i + koff + r);
                const float  v   = acc[i][t][r] + bv;
                if constexpr (WF32)  Cf[row * Ndim + col]  = v;
                if constexpr (WBF16) Cbf[row * Ndim + col] = (__bf16)v;
            }
        }
    }
}

// ---------------------------------------------------------------------------
// Flash attention. Block = 256 thr = 8 waves sharing one (b,h); each wave owns
// a 16-row Q tile. Double-buffered LDS: TDM DMAs the NEXT K chunk while the
// current one is computed; V staged transposed with regs-early/store-late.
// Distinct fragment registers so ds_load_b128s batch and waits stay partial.
// ---------------------------------------------------------------------------
__global__ __launch_bounds__(256) void attn_flash_bf16(
    const __bf16* __restrict__ Qb, const __bf16* __restrict__ Kb,
    const __bf16* __restrict__ Vb, __bf16* __restrict__ Ob) {
    __shared__ __bf16 sK[2][32][HDIM];   // [buf][key][d]
    __shared__ __bf16 sVt[2][HDIM][32];  // [buf][d][key]

    const int tid   = threadIdx.x;
    const int wave  = tid >> 5;
    const int lane  = tid & 31;
    const int bh    = blockIdx.x >> 5;
    const int qblk  = blockIdx.x & 31;
    const int b     = bh >> 4;
    const int h     = bh & 15;
    const int qbase = qblk * 128 + wave * 16;
    const int lrow  = lane & 15;
    const int koff  = (lane >> 4) << 3;

    const __bf16* Qrow  = Qb + ((size_t)b * NQ + qbase + lrow) * EMB + h * HDIM;
    const __bf16* Kbase = Kb + (size_t)b * MK * EMB + h * HDIM;
    const __bf16* Vbase = Vb + (size_t)b * MK * EMB + h * HDIM;

    FragBF qf0, qf1;  // Q^T fragments (B operand): lane = q-row
    qf0.u[0] = *(const uint4*)(Qrow + koff);
    qf0.u[1] = *(const uint4*)(Qrow + 16 + koff);
    qf1.u[0] = *(const uint4*)(Qrow + 32 + koff);
    qf1.u[1] = *(const uint4*)(Qrow + 48 + koff);

    v8f acc[4] = {{}, {}, {}, {}};
    float m_run = -1e30f;
    float l_run = 0.0f;

    const int srow = tid >> 3;       // 0..31
    const int sd   = (tid & 7) * 8;  // 0..56

    // ---- prologue: stage chunk 0 into buffer 0 ----
#if HAVE_TDM
    if (wave == 0)
        tdm_load_2d_bf16(Kbase, (unsigned)(uintptr_t)(&sK[0][0][0]), 32, HDIM, EMB);
#else
    {
        uint4 kv = *(const uint4*)(Kbase + (size_t)srow * EMB + sd);
        *(uint4*)(&sK[0][srow][sd]) = kv;
    }
#endif
    {
        Pack8 vv;
        vv.u = *(const uint4*)(Vbase + (size_t)srow * EMB + sd);
#pragma unroll
        for (int j = 0; j < 8; ++j) sVt[0][sd + j][srow] = vv.e[j];
    }
#if HAVE_TDM
    if (wave == 0) __builtin_amdgcn_s_wait_tensorcnt(0);
#endif
    __syncthreads();

    const int niter = MK / 32;
    for (int it = 0; it < niter; ++it) {
        const int  buf      = it & 1;
        const bool has_next = (it + 1) < niter;
        Pack8 vv = {};
#if !HAVE_TDM
        uint4 kv = {};
#endif
        if (has_next) {
            const size_t nb = (size_t)((it + 1) * 32 + srow) * EMB + sd;
            vv.u = *(const uint4*)(Vbase + nb);  // issue early
#if HAVE_TDM
            if (wave == 0)  // DMA next K chunk into the other buffer
                tdm_load_2d_bf16(Kbase + (size_t)(it + 1) * 32 * EMB,
                                 (unsigned)(uintptr_t)(&sK[buf ^ 1][0][0]), 32, HDIM,
                                 EMB);
#else
            kv = *(const uint4*)(Kbase + nb);
#endif
        }

        // ---- S^T = K_chunk . Q^T (all LDS loads issued before the WMMAs) ----
        FragBF kf0, kf1, kf2, kf3;
        kf0.u[0] = *(const uint4*)(&sK[buf][lrow][koff]);
        kf0.u[1] = *(const uint4*)(&sK[buf][lrow][16 + koff]);
        kf1.u[0] = *(const uint4*)(&sK[buf][lrow][32 + koff]);
        kf1.u[1] = *(const uint4*)(&sK[buf][lrow][48 + koff]);
        kf2.u[0] = *(const uint4*)(&sK[buf][16 + lrow][koff]);
        kf2.u[1] = *(const uint4*)(&sK[buf][16 + lrow][16 + koff]);
        kf3.u[0] = *(const uint4*)(&sK[buf][16 + lrow][32 + koff]);
        kf3.u[1] = *(const uint4*)(&sK[buf][16 + lrow][48 + koff]);
        v8f s_lo = {}, s_hi = {};
        s_lo = wmma_bf16(kf0.v, qf0.v, s_lo);
        s_lo = wmma_bf16(kf1.v, qf1.v, s_lo);
        s_hi = wmma_bf16(kf2.v, qf0.v, s_hi);
        s_hi = wmma_bf16(kf3.v, qf1.v, s_hi);

        // ---- V fragments for this chunk: issue loads before the softmax ----
        FragBF vf[4];
#pragma unroll
        for (int t = 0; t < 4; ++t) {
            vf[t].u[0] = *(const uint4*)(&sVt[buf][16 * t + lrow][koff]);
            vf[t].u[1] = *(const uint4*)(&sVt[buf][16 * t + lrow][16 + koff]);
        }

        // ---- online softmax: lane holds 16 scores for q-row lrow ----
        float mloc = -1e30f;
#pragma unroll
        for (int e = 0; e < 8; ++e) {
            s_lo[e] *= ATT_SCALE;
            s_hi[e] *= ATT_SCALE;
            mloc = fmaxf(mloc, fmaxf(s_lo[e], s_hi[e]));
        }
        mloc = fmaxf(mloc, __shfl_xor(mloc, 16, 32));
        const float mnew = fmaxf(m_run, mloc);
        const float corr = __expf(m_run - mnew);
        float p_lo[8], p_hi[8];
        float ssum = 0.0f;
#pragma unroll
        for (int e = 0; e < 8; ++e) {
            p_lo[e] = __expf(s_lo[e] - mnew);
            p_hi[e] = __expf(s_hi[e] - mnew);
            ssum += p_lo[e] + p_hi[e];
        }
        ssum += __shfl_xor(ssum, 16, 32);
        l_run = l_run * corr + ssum;
        m_run = mnew;

        float fac[8];
#pragma unroll
        for (int r = 0; r < 8; ++r) fac[r] = __shfl(corr, koff + r, 32);
#pragma unroll
        for (int t = 0; t < 4; ++t)
#pragma unroll
            for (int r = 0; r < 8; ++r) acc[t][r] *= fac[r];

        // ---- P fragment (already in A-fragment layout) ----
        union { v16bf v; __bf16 e[16]; } pf;
#pragma unroll
        for (int e = 0; e < 8; ++e) {
            pf.e[e]     = (__bf16)p_lo[e];
            pf.e[8 + e] = (__bf16)p_hi[e];
        }
        // ---- O += P . V ----
#pragma unroll
        for (int t = 0; t < 4; ++t) acc[t] = wmma_bf16(pf.v, vf[t].v, acc[t]);

        // ---- fill the other buffer; TDM completes before the barrier ----
        if (has_next) {
#pragma unroll
            for (int j = 0; j < 8; ++j) sVt[buf ^ 1][sd + j][srow] = vv.e[j];
#if HAVE_TDM
            if (wave == 0) __builtin_amdgcn_s_wait_tensorcnt(0);
#else
            *(uint4*)(&sK[buf ^ 1][srow][sd]) = kv;
#endif
        }
        __syncthreads();
    }

    const float linv = 1.0f / l_run;
    float fac[8];
#pragma unroll
    for (int r = 0; r < 8; ++r) fac[r] = __shfl(linv, koff + r, 32);
    __bf16* Orow = Ob + ((size_t)b * NQ + qbase) * EMB + h * HDIM;
#pragma unroll
    for (int t = 0; t < 4; ++t)
#pragma unroll
        for (int r = 0; r < 8; ++r)
            Orow[(size_t)(koff + r) * EMB + 16 * t + lrow] = (__bf16)(acc[t][r] * fac[r]);
}

// ---------------------------------------------------------------------------
extern "C" void kernel_launch(void* const* d_in, const int* in_sizes, int n_in,
                              void* d_out, int out_size, void* d_ws, size_t ws_size,
                              hipStream_t stream) {
    const float* tokens  = (const float*)d_in[0];
    const float* context = (const float*)d_in[1];
    const float* Wq      = (const float*)d_in[2];
    const float* Wk      = (const float*)d_in[3];
    const float* Wv      = (const float*)d_in[4];
    const float* Wo      = (const float*)d_in[5];
    const float* bo      = (const float*)d_in[6];
    float*       out     = (float*)d_out;

    const size_t nTok = (size_t)BATCH * NQ * EMB;
    const size_t nCtx = (size_t)BATCH * MK * CTX;
    const size_t nWq  = (size_t)EMB * EMB;
    const size_t nWk  = (size_t)CTX * EMB;
    const size_t nWv  = (size_t)CTX * EMB;
    const size_t nWo  = (size_t)EMB * EMB;
    const size_t nK   = (size_t)BATCH * MK * EMB;

    char*  ws  = (char*)d_ws;
    size_t off = 0;
    auto carve = [&](size_t elems) -> __bf16* {
        __bf16* p = (__bf16*)(ws + off);
        off = (off + elems * sizeof(__bf16) + 255) & ~(size_t)255;
        return p;
    };
    __bf16* tok_bf = carve(nTok);
    __bf16* ctx_bf = carve(nCtx);
    __bf16* Wq_bf  = carve(nWq);
    __bf16* Wk_bf  = carve(nWk);
    __bf16* Wv_bf  = carve(nWv);
    __bf16* Wo_bf  = carve(nWo);
    __bf16* Q_bf   = carve(nTok);
    __bf16* K_bf   = carve(nK);
    __bf16* V_bf   = carve(nK);
    __bf16* AO_bf  = carve(nTok);
    (void)ws_size;  // ~131 MB required

    auto cast = [&](const float* s, __bf16* d, size_t n) {
        int blocks = (int)((n / 4 + 255) / 256);
        cast_f32_bf16<<<blocks, 256, 0, stream>>>(s, d, (int)n);
    };
    cast(tokens, tok_bf, nTok);
    cast(context, ctx_bf, nCtx);
    cast(Wq, Wq_bf, nWq);
    cast(Wk, Wk_bf, nWk);
    cast(Wv, Wv_bf, nWv);
    cast(Wo, Wo_bf, nWo);

    gemm_bf16_wmma<false, true, false>
        <<<dim3(EMB / 128, (BATCH * NQ) / 128), 256, 0, stream>>>(
            tok_bf, Wq_bf, nullptr, nullptr, Q_bf, BATCH * NQ, EMB, EMB);
    gemm_bf16_wmma<false, true, false>
        <<<dim3(EMB / 128, (BATCH * MK) / 128), 256, 0, stream>>>(
            ctx_bf, Wk_bf, nullptr, nullptr, K_bf, BATCH * MK, EMB, CTX);
    gemm_bf16_wmma<false, true, false>
        <<<dim3(EMB / 128, (BATCH * MK) / 128), 256, 0, stream>>>(
            ctx_bf, Wv_bf, nullptr, nullptr, V_bf, BATCH * MK, EMB, CTX);

    attn_flash_bf16<<<dim3(BATCH * NHEAD * (NQ / 128)), 256, 0, stream>>>(
        Q_bf, K_bf, V_bf, AO_bf);

    gemm_bf16_wmma<true, false, true>
        <<<dim3(EMB / 128, (BATCH * NQ) / 128), 256, 0, stream>>>(
            AO_bf, Wo_bf, bo, out, nullptr, BATCH * NQ, EMB, EMB);
    (void)in_sizes; (void)n_in; (void)out_size;
}
